// MultiHeadAttention_14568529068339
// MI455X (gfx1250) — matrix operand order repaired and checked by reading, not compiled, not run on hardware
//
#include <hip/hip_runtime.h>

typedef __bf16 bf16_t;
typedef __attribute__((ext_vector_type(16))) __bf16 v16bf;
typedef __attribute__((ext_vector_type(8)))  __bf16 v8bf;
typedef __attribute__((ext_vector_type(8)))  float  v8f;
typedef __attribute__((ext_vector_type(16))) float  v16f;

#define B_  8
#define T_  1024
#define E_  768
#define H_  12
#define DH_ 64
#define M_  (B_ * T_)      // 8192

// ---------------------------------------------------------------------------
// WMMA wrapper: D = A(16x32 bf16) * B(32x16 bf16) + C(16x16 f32)
// ---------------------------------------------------------------------------
__device__ __forceinline__ v8f wmma_bf16(v16bf a, v16bf b, v8f c) {
  return __builtin_amdgcn_wmma_f32_16x16x32_bf16(
      /*neg_a=*/false, a, /*neg_b=*/false, b,
      /*c_mod=*/(short)0, c, /*reuse_a=*/false, /*reuse_b=*/false);
}

// A fragment (16x32, 16-bit): lane l (ln=l&15, lh=l>>4) holds row M=ln,
//   VGPR0-3: K = lh*8 .. lh*8+7 ; VGPR4-7: K = 16+lh*8 .. 16+lh*8+7
__device__ __forceinline__ v16bf afrag_f32(const float* row, int k0, int lh) {
  const v8f lo = *(const v8f*)(row + k0 + lh * 8);
  const v8f hi = *(const v8f*)(row + k0 + 16 + lh * 8);
  v16bf a;
#pragma unroll
  for (int i = 0; i < 8; ++i) { a[i] = (__bf16)lo[i]; a[i + 8] = (__bf16)hi[i]; }
  return a;
}
__device__ __forceinline__ v16bf afrag_bf(const bf16_t* row, int k0, int lh) {
  const v8bf lo = *(const v8bf*)(row + k0 + lh * 8);
  const v8bf hi = *(const v8bf*)(row + k0 + 16 + lh * 8);
  v16bf a;
#pragma unroll
  for (int i = 0; i < 8; ++i) { a[i] = lo[i]; a[i + 8] = hi[i]; }
  return a;
}

// B fragment (32x16, 16-bit): lane l holds column N=ln,
// elements 0-7: K = lh*8 .. lh*8+7 ; elements 8-15: K = 16+lh*8 .. 16+lh*8+7
__device__ __forceinline__ v16bf bfrag_f32(const float* row, int k0, int lh) {
  const v8f lo = *(const v8f*)(row + k0 + lh * 8);
  const v8f hi = *(const v8f*)(row + k0 + 16 + lh * 8);
  v16bf b;
#pragma unroll
  for (int i = 0; i < 8; ++i) { b[i] = (__bf16)lo[i]; b[i + 8] = (__bf16)hi[i]; }
  return b;
}
__device__ __forceinline__ v16bf bfrag_bf(const bf16_t* row, int k0, int lh) {
  const v8bf lo = *(const v8bf*)(row + k0 + lh * 8);
  const v8bf hi = *(const v8bf*)(row + k0 + 16 + lh * 8);
  v16bf b;
#pragma unroll
  for (int i = 0; i < 8; ++i) { b[i] = lo[i]; b[i + 8] = hi[i]; }
  return b;
}

__device__ __forceinline__ float redmax16(float v) {
  v = fmaxf(v, __shfl_xor(v, 1, 32));
  v = fmaxf(v, __shfl_xor(v, 2, 32));
  v = fmaxf(v, __shfl_xor(v, 4, 32));
  v = fmaxf(v, __shfl_xor(v, 8, 32));
  return v;
}
__device__ __forceinline__ float redsum16(float v) {
  v += __shfl_xor(v, 1, 32);
  v += __shfl_xor(v, 2, 32);
  v += __shfl_xor(v, 4, 32);
  v += __shfl_xor(v, 8, 32);
  return v;
}

// ---------------------------------------------------------------------------
// Projection GEMM: C[8192,768] = X(f32) @ W(f32)^T + bias, stored as bf16 in
// head layout. vt_mode==0 -> dst[B,H,T,DH]; vt_mode==1 -> dst[B,H,DH,T].
// One wave = 64x64 tile (4x4 WMMA fragments), K-loop step 32.
// ---------------------------------------------------------------------------
__global__ void __launch_bounds__(128)
gemm_proj(const float* __restrict__ X, const float* __restrict__ W,
          const float* __restrict__ bias, bf16_t* __restrict__ dst,
          int vt_mode) {
  const int lane = threadIdx.x & 31;
  const int ln = lane & 15, lh = lane >> 4;
  const int wid = (blockIdx.x * blockDim.x + threadIdx.x) >> 5;  // 0..1535
  const int mt = wid / (E_ / 64), nt = wid % (E_ / 64);
  const int m0 = mt * 64, n0 = nt * 64;

  const v8f zv = {0.f, 0.f, 0.f, 0.f, 0.f, 0.f, 0.f, 0.f};
  v8f acc[4][4];
#pragma unroll
  for (int i = 0; i < 4; ++i)
#pragma unroll
    for (int j = 0; j < 4; ++j) acc[i][j] = zv;

  for (int kb = 0; kb < E_; kb += 32) {
    v16bf af[4], bfr[4];
#pragma unroll
    for (int i = 0; i < 4; ++i)
      af[i] = afrag_f32(X + (size_t)(m0 + 16 * i + ln) * E_, kb, lh);
#pragma unroll
    for (int j = 0; j < 4; ++j)
      bfr[j] = bfrag_f32(W + (size_t)(n0 + 16 * j + ln) * E_, kb, lh);
#pragma unroll
    for (int i = 0; i < 4; ++i)
#pragma unroll
      for (int j = 0; j < 4; ++j) acc[i][j] = wmma_bf16(af[i], bfr[j], acc[i][j]);
  }

#pragma unroll
  for (int i = 0; i < 4; ++i) {
#pragma unroll
    for (int j = 0; j < 4; ++j) {
#pragma unroll
      for (int r = 0; r < 8; ++r) {
        const int m = m0 + 16 * i + 8 * lh + r;   // C row = r + 8*lh
        const int n = n0 + 16 * j + ln;           // C col = ln
        const float val = acc[i][j][r] + bias[n];
        const int b = m >> 10, t = m & (T_ - 1);
        const int h = n >> 6, d = n & (DH_ - 1);
        const size_t idx = vt_mode
            ? ((size_t)((b * H_ + h) * DH_ + d)) * T_ + t
            : ((size_t)((b * H_ + h) * T_ + t)) * DH_ + d;
        dst[idx] = (bf16_t)val;
      }
    }
  }
}

// ---------------------------------------------------------------------------
// Output GEMM: out[8192,768](f32) = merged(bf16) @ Wo(f32)^T + bo
// ---------------------------------------------------------------------------
__global__ void __launch_bounds__(128)
gemm_out(const bf16_t* __restrict__ X, const float* __restrict__ W,
         const float* __restrict__ bias, float* __restrict__ out) {
  const int lane = threadIdx.x & 31;
  const int ln = lane & 15, lh = lane >> 4;
  const int wid = (blockIdx.x * blockDim.x + threadIdx.x) >> 5;
  const int mt = wid / (E_ / 64), nt = wid % (E_ / 64);
  const int m0 = mt * 64, n0 = nt * 64;

  const v8f zv = {0.f, 0.f, 0.f, 0.f, 0.f, 0.f, 0.f, 0.f};
  v8f acc[4][4];
#pragma unroll
  for (int i = 0; i < 4; ++i)
#pragma unroll
    for (int j = 0; j < 4; ++j) acc[i][j] = zv;

  for (int kb = 0; kb < E_; kb += 32) {
    v16bf af[4], bfr[4];
#pragma unroll
    for (int i = 0; i < 4; ++i)
      af[i] = afrag_bf(X + (size_t)(m0 + 16 * i + ln) * E_, kb, lh);
#pragma unroll
    for (int j = 0; j < 4; ++j)
      bfr[j] = bfrag_f32(W + (size_t)(n0 + 16 * j + ln) * E_, kb, lh);
#pragma unroll
    for (int i = 0; i < 4; ++i)
#pragma unroll
      for (int j = 0; j < 4; ++j) acc[i][j] = wmma_bf16(af[i], bfr[j], acc[i][j]);
  }

#pragma unroll
  for (int i = 0; i < 4; ++i)
#pragma unroll
    for (int j = 0; j < 4; ++j)
#pragma unroll
      for (int r = 0; r < 8; ++r) {
        const int m = m0 + 16 * i + 8 * lh + r;
        const int n = n0 + 16 * j + ln;
        out[(size_t)m * E_ + n] = acc[i][j][r] + bias[n];
      }
}

// ---------------------------------------------------------------------------
// Flash-style attention. One wave per (b, h, 16-query tile).
// Qh,Kh: [B,H,T,DH] bf16.  Vt: [B,H,DH,T] bf16.  merged: [B,T,E] bf16.
// Streaming softmax over 32-key chunks; masked keys get score 1e-9 (faithful).
// ---------------------------------------------------------------------------
__global__ void __launch_bounds__(128)
attn_kernel(const bf16_t* __restrict__ Qh, const bf16_t* __restrict__ Kh,
            const bf16_t* __restrict__ Vt, const int* __restrict__ msk,
            bf16_t* __restrict__ merged) {
  __shared__ __attribute__((aligned(32))) bf16_t ptile[4][16 * 32];
  const int lane = threadIdx.x & 31;
  const int ln = lane & 15, lh = lane >> 4;
  const int wslot = threadIdx.x >> 5;
  const int wid = (blockIdx.x * blockDim.x + threadIdx.x) >> 5;  // 0..6143
  const int qt = wid & 63;          // T/16 = 64 query tiles
  const int bh = wid >> 6;          // 0..95
  const int b = bh / H_, h = bh % H_;
  const int q0 = qt * 16;

  const bf16_t* Qbh = Qh + (size_t)bh * T_ * DH_;
  const bf16_t* Kbh = Kh + (size_t)bh * T_ * DH_;
  const bf16_t* Vbh = Vt + (size_t)bh * DH_ * T_;
  const int* mrow = msk + b * T_;
  bf16_t* lds = ptile[wslot];

  // Q tile: 16x64 = two A fragments (d 0..31, 32..63), loaded once.
  const v16bf qa0 = afrag_bf(Qbh + (size_t)(q0 + ln) * DH_, 0, lh);
  const v16bf qa1 = afrag_bf(Qbh + (size_t)(q0 + ln) * DH_, 32, lh);

  const v8f zv = {0.f, 0.f, 0.f, 0.f, 0.f, 0.f, 0.f, 0.f};
  v8f acc[4];  // O accumulator: 16 x 64 f32 (dh subtiles of 16)
#pragma unroll
  for (int j = 0; j < 4; ++j) acc[j] = zv;
  float mrun[8], lrun[8];
#pragma unroll
  for (int r = 0; r < 8; ++r) { mrun[r] = -INFINITY; lrun[r] = 0.f; }

  for (int kc = 0; kc < T_; kc += 32) {
    // Scores for 32 keys: two 16x16 fragments, each = 2 WMMAs over d.
    v8f sa = zv, sb = zv;
    sa = wmma_bf16(qa0, bfrag_bf(Kbh + (size_t)(kc + ln) * DH_, 0, lh), sa);
    sa = wmma_bf16(qa1, bfrag_bf(Kbh + (size_t)(kc + ln) * DH_, 32, lh), sa);
    sb = wmma_bf16(qa0, bfrag_bf(Kbh + (size_t)(kc + 16 + ln) * DH_, 0, lh), sb);
    sb = wmma_bf16(qa1, bfrag_bf(Kbh + (size_t)(kc + 16 + ln) * DH_, 32, lh), sb);

    const bool va = mrow[kc + ln] != 0;
    const bool vb = mrow[kc + 16 + ln] != 0;

#pragma unroll
    for (int r = 0; r < 8; ++r) {
      float s0 = va ? sa[r] * 0.125f : 1e-9f;   // 1/sqrt(64); masked -> 1e-9
      float s1 = vb ? sb[r] * 0.125f : 1e-9f;
      const float mx = redmax16(fmaxf(s0, s1));
      const float nm = fmaxf(mrun[r], mx);
      const float corr = __expf(mrun[r] - nm);
      mrun[r] = nm;
      const float p0 = __expf(s0 - nm);
      const float p1 = __expf(s1 - nm);
      lrun[r] = lrun[r] * corr + redsum16(p0 + p1);
#pragma unroll
      for (int j = 0; j < 4; ++j) acc[j][r] *= corr;
      const int row = 8 * lh + r;               // C-layout row for this lane
      lds[row * 32 + ln] = (bf16_t)p0;
      lds[row * 32 + 16 + ln] = (bf16_t)p1;
    }

    // Reload P (16x32) in A-fragment layout (wave-synchronous; LDS in-order).
    v16bf pf;
#pragma unroll
    for (int i = 0; i < 8; ++i) {
      pf[i]     = lds[ln * 32 + lh * 8 + i];
      pf[i + 8] = lds[ln * 32 + 16 + lh * 8 + i];
    }

    // O += P(16x32) * V(32x16) per dh-subtile.
#pragma unroll
    for (int j = 0; j < 4; ++j) {
      const v16bf vb_ = bfrag_bf(Vbh + (size_t)(16 * j + ln) * T_ + kc, 0, lh);
      acc[j] = wmma_bf16(pf, vb_, acc[j]);
    }
  }

#pragma unroll
  for (int r = 0; r < 8; ++r) {
    const float inv = 1.0f / lrun[r];
    const int t = q0 + 8 * lh + r;
#pragma unroll
    for (int j = 0; j < 4; ++j)
      merged[(size_t)(b * T_ + t) * E_ + h * DH_ + 16 * j + ln] =
          (bf16_t)(acc[j][r] * inv);
  }
}

// ---------------------------------------------------------------------------
extern "C" void kernel_launch(void* const* d_in, const int* in_sizes, int n_in,
                              void* d_out, int out_size, void* d_ws,
                              size_t ws_size, hipStream_t stream) {
  (void)in_sizes; (void)n_in; (void)out_size; (void)ws_size;
  const float* q  = (const float*)d_in[0];
  const float* k  = (const float*)d_in[1];
  const float* v  = (const float*)d_in[2];
  const int* mask = (const int*)d_in[3];
  // d_in[4] = causal_mask: unused by the reference module
  const float* Wq = (const float*)d_in[5];
  const float* bq = (const float*)d_in[6];
  const float* Wk = (const float*)d_in[7];
  const float* bk = (const float*)d_in[8];
  const float* Wv = (const float*)d_in[9];
  const float* bv = (const float*)d_in[10];
  const float* Wo = (const float*)d_in[11];
  const float* bo = (const float*)d_in[12];

  char* ws = (char*)d_ws;
  const size_t NTE = (size_t)M_ * E_;           // 6,291,456 elements
  bf16_t* Qh     = (bf16_t*)(ws);               // [B,H,T,DH]
  bf16_t* Kh     = (bf16_t*)(ws + NTE * 2);     // [B,H,T,DH]
  bf16_t* Vt     = (bf16_t*)(ws + NTE * 4);     // [B,H,DH,T]
  bf16_t* merged = (bf16_t*)(ws + NTE * 6);     // [B,T,E]   (48 MiB total)

  const dim3 blk(128);
  // 128 M-tiles x 12 N-tiles = 1536 wave-tiles = 384 blocks of 4 waves
  gemm_proj<<<384, blk, 0, stream>>>(q, Wq, bq, Qh, 0);
  gemm_proj<<<384, blk, 0, stream>>>(k, Wk, bk, Kh, 0);
  gemm_proj<<<384, blk, 0, stream>>>(v, Wv, bv, Vt, 1);
  // 8*12*64 = 6144 (b,h,qtile) waves = 1536 blocks of 4 waves
  attn_kernel<<<1536, blk, 0, stream>>>(Qh, Kh, Vt, mask, merged);
  gemm_out<<<384, blk, 0, stream>>>(merged, Wo, bo, (float*)d_out);
}
